// ReconstructionHGTModel_42872363548696
// MI455X (gfx1250) — matrix operand order repaired and checked by reading, not compiled, not run on hardware
//
#include <hip/hip_runtime.h>
#include <hip/hip_bf16.h>
#include <math.h>

typedef __attribute__((ext_vector_type(16))) _Float16 v16h;
typedef __attribute__((ext_vector_type(8)))  _Float16 v8h;
typedef __attribute__((ext_vector_type(8)))  float    v8f;

#define NROOM   100000
#define NPORTAL 100000
#define NEDGE   800000
#define NCAND   200000
#define HIDDEN  128
#define CHUNK   50000

// ---------------------------------------------------------------- utilities
__device__ __forceinline__ float wave_sum(float v) {
#pragma unroll
  for (int o = 16; o > 0; o >>= 1) v += __shfl_xor(v, o, 32);
  return v;
}
__device__ __forceinline__ unsigned fenc(float f) {
  unsigned u = __float_as_uint(f);
  return (u & 0x80000000u) ? ~u : (u | 0x80000000u);
}
__device__ __forceinline__ float fdec(unsigned u) {
  u = (u & 0x80000000u) ? (u & 0x7FFFFFFFu) : ~u;
  return __uint_as_float(u);
}

// WMMA B-fragment swizzle: element (k,n) -> [kt][nt][lane][half]
// lane: N = lane%16, +16 if (k%32)>=16 ; half: K%16
__device__ __forceinline__ long bswz(int k, int n, int NTG) {
  int kt = k >> 5, kk = k & 31;
  int nt = n >> 4;
  int l2 = (n & 15) | (kk & 16);
  int h2 = kk & 15;
  return ((long)(kt * NTG + nt) << 9) + (l2 << 4) + h2;
}

__global__ void fill_u32_kernel(unsigned* __restrict__ p, unsigned v, long n) {
  long i = (long)blockIdx.x * 256 + threadIdx.x;
  if (i < n) p[i] = v;
}

// op 0: identity, 1: exact gelu
__global__ void convert_kernel(const float* __restrict__ in, _Float16* __restrict__ out,
                               long n, int op) {
  long i = (long)blockIdx.x * 256 + threadIdx.x;
  if (i >= n) return;
  float v = in[i];
  if (op == 1) v = 0.5f * v * (1.0f + erff(v * 0.70710678118654752f));
  out[i] = (_Float16)v;
}

// ---------------------------------------------------------------- weight prep
// Bt (pre-swizzled) <- W[n*K + k]
__global__ void prep_weight_plain(const float* __restrict__ W, _Float16* __restrict__ Bt,
                                  int K, int N) {
  int idx = blockIdx.x * 256 + threadIdx.x;
  if (idx >= K * N) return;
  int k = idx / N, n = idx - k * N;
  Bt[bswz(k, n, N >> 4)] = (_Float16)W[n * K + k];
}

// fused: out col o=h*32+e over 128; Bt swz[k,o] = sum_d W[(h*32+d)*128+k]*R[(h*32+d)*32+e]
__global__ void prep_weight_fused(const float* __restrict__ W, const float* __restrict__ b,
                                  const float* __restrict__ R,
                                  _Float16* __restrict__ Bt, float* __restrict__ fb) {
  int idx = blockIdx.x * 256 + threadIdx.x;
  if (idx >= 128 * 128) return;
  int k = idx >> 7, o = idx & 127;
  int h4 = o >> 5, e = o & 31;
  float s = 0.f;
#pragma unroll 8
  for (int d = 0; d < 32; d++)
    s += W[(h4 * 32 + d) * 128 + k] * R[(h4 * 32 + d) * 32 + e];
  Bt[bswz(k, o, 8)] = (_Float16)s;
  if (k == 0) {
    float sb = 0.f;
    for (int d = 0; d < 32; d++)
      sb += b[h4 * 32 + d] * R[(h4 * 32 + d) * 32 + e];
    fb[o] = sb;
  }
}

__global__ void prep_weight_pair(const float* __restrict__ lpw, const float* __restrict__ ewpw,
                                 const float* __restrict__ lpb, const float* __restrict__ ewpb,
                                 _Float16* __restrict__ Bt, float* __restrict__ bcat) {
  int idx = blockIdx.x * 256 + threadIdx.x;
  if (idx >= 256 * 256) return;
  int k = idx >> 8, n = idx & 255;
  float v = (n < 128) ? lpw[n * 256 + k] : ewpw[(n - 128) * 256 + k];
  Bt[bswz(k, n, 16)] = (_Float16)v;
  if (k == 0) bcat[n] = (n < 128) ? lpb[n] : ewpb[n - 128];
}

// ---------------------------------------------------------------- encoder
__global__ void encode_kernel(const float* __restrict__ x, const float* __restrict__ w,
                              const float* __restrict__ b, float* __restrict__ h, int n) {
  long idx = (long)blockIdx.x * 256 + threadIdx.x;
  if (idx >= (long)n * 128) return;
  int r = (int)(idx >> 7), j = (int)(idx & 127);
  h[idx] = x[r] * w[j] + b[j];
}

// ---------------------------------------------------------------- WMMA GEMM
// C[M x N] = A[M x KT*32](f16) @ Bt(pre-swizzled, KT*32 x N)(f16) + bias[N]
// mode 0: store; mode 1: relu(beta*v + (1-beta)*hold), beta = sigmoid(*skip_p); mode 2: relu
// Whole B panel for this blockIdx.y N-chunk staged in LDS once (no per-K barriers).
template <int NT, int KT>
__global__ __launch_bounds__(256) void gemm_f16_kernel(
    const _Float16* __restrict__ A, const _Float16* __restrict__ Bt,
    const float* __restrict__ bias, float* __restrict__ C,
    int M, int N, int mode,
    const float* __restrict__ hold, const float* __restrict__ skip_p) {
  __shared__ alignas(32) _Float16 lds[NT * KT * 512];
  const int tid   = threadIdx.x;
  const int wave  = tid >> 5;
  const int lane  = tid & 31;
  const int laneM = lane & 15;
  const int hi    = lane >> 4;
  const int NTG   = N >> 4;                 // total 16-col tiles in B
  const int nt0   = blockIdx.y * NT;        // first tile of this N-chunk
  const int n0    = nt0 * 16;
  const int m0    = blockIdx.x * 128 + wave * 16;
  const int row   = m0 + laneM;
  const bool rowOK = row < M;

  // stage entire (pre-swizzled) B panel: contiguous per K-tile
#pragma unroll
  for (int i = tid; i < NT * KT * 64; i += 256) {   // 64 v8h chunks per frag-group
    int kt  = i / (NT * 64);
    int rem = i - kt * (NT * 64);
    ((v8h*)lds)[kt * NT * 64 + rem] =
        ((const v8h*)(Bt + (((long)kt * NTG + nt0) << 9)))[rem];
  }
  __syncthreads();

  v8f acc[NT] = {};
#pragma unroll
  for (int kt = 0; kt < KT; kt++) {
    // A fragment (matches B swizzle): lane<16 -> K 0..15 ; lane>=16 -> K 16..31
    union { v16h v; v8h h[2]; } ua;
    if (rowOK) {
      const _Float16* base = A + (long)row * (KT * 32) + kt * 32 + (hi ? 16 : 0);
      ua.h[0] = *(const v8h*)(base);
      ua.h[1] = *(const v8h*)(base + 8);
    } else {
      v8h z = {};
      ua.h[0] = z; ua.h[1] = z;
    }
    v16h a = ua.v;
#pragma unroll
    for (int nt = 0; nt < NT; nt++) {
      v16h b = *(const v16h*)(&lds[(kt * NT + nt) * 512 + lane * 16]);
      acc[nt] = __builtin_amdgcn_wmma_f32_16x16x32_f16(
          false, a, false, b, (short)0, acc[nt], false, false);
    }
  }

  float beta = 0.f;
  if (mode == 1) beta = 1.f / (1.f + __expf(-skip_p[0]));
#pragma unroll
  for (int nt = 0; nt < NT; nt++) {
    int col = n0 + nt * 16 + laneM;
    float bv = bias ? bias[col] : 0.f;
#pragma unroll
    for (int r = 0; r < 8; r++) {
      int rr = m0 + r + hi * 8;
      if (rr >= M) continue;
      float v = acc[nt][r] + bv;
      if (mode == 1) {
        float o = beta * v + (1.f - beta) * hold[(long)rr * N + col];
        v = fmaxf(o, 0.f);
      } else if (mode == 2) {
        v = fmaxf(v, 0.f);
      }
      C[(long)rr * N + col] = v;
    }
  }
}

// ---------------------------------------------------------------- edge phase
__global__ __launch_bounds__(256) void edge_logits_kernel(
    const float* __restrict__ q, const float* __restrict__ krel,
    const int* __restrict__ ei, const float* __restrict__ prel,
    float* __restrict__ logits, unsigned* __restrict__ nmax, int ne) {
  int e = blockIdx.x * 8 + (threadIdx.x >> 5);
  int lane = threadIdx.x & 31;
  if (e >= ne) return;
  int src = ei[e], dst = ei[ne + e];
  const float* qp = q + (long)dst * 128;
  const float* kp = krel + (long)src * 128;
#pragma unroll
  for (int j = 0; j < 4; j++) {
    float p = qp[j * 32 + lane] * kp[j * 32 + lane];
    p = wave_sum(p);
    if (lane == 0) {
      float lg = p * prel[j] * 0.17677669529663687f;  // 1/sqrt(32)
      logits[(long)e * 4 + j] = lg;
      atomicMax(&nmax[dst * 4 + j], fenc(lg));
    }
  }
}

__global__ void edge_exp_kernel(float* __restrict__ logits, const unsigned* __restrict__ nmax,
                                float* __restrict__ denom, const int* __restrict__ ei, int ne) {
  long idx = (long)blockIdx.x * 256 + threadIdx.x;
  if (idx >= (long)ne * 4) return;
  int e = (int)(idx >> 2), j = (int)(idx & 3);
  int dst = ei[ne + e];
  float m = fdec(nmax[dst * 4 + j]);
  float ev = __expf(logits[idx] - m);
  logits[idx] = ev;
  atomicAdd(&denom[dst * 4 + j], ev);
}

__global__ __launch_bounds__(256) void edge_aggr_kernel(
    const float* __restrict__ expv, const float* __restrict__ denom,
    const float* __restrict__ vrel, const int* __restrict__ ei,
    float* __restrict__ aggr, int ne) {
  int e = blockIdx.x * 8 + (threadIdx.x >> 5);
  int lane = threadIdx.x & 31;
  if (e >= ne) return;
  int src = ei[e], dst = ei[ne + e];
#pragma unroll
  for (int j = 0; j < 4; j++) {
    float alpha = expv[(long)e * 4 + j] / denom[dst * 4 + j];
    atomicAdd(&aggr[(long)dst * 128 + j * 32 + lane],
              vrel[(long)src * 128 + j * 32 + lane] * alpha);
  }
}

// ---------------------------------------------------------------- pooling / heads
__global__ __launch_bounds__(256) void pool_kernel(
    const float* __restrict__ h, const float* __restrict__ attn_w,
    const float* __restrict__ attn_b, float* __restrict__ g, int n) {
  int wave = threadIdx.x >> 5, lane = threadIdx.x & 31;
  float a0 = 0, a1 = 0, a2 = 0, a3 = 0;
  float w0 = attn_w[lane], w1 = attn_w[32 + lane], w2 = attn_w[64 + lane], w3 = attn_w[96 + lane];
  float b = attn_b[0];
  for (int r = blockIdx.x * 8 + wave; r < n; r += gridDim.x * 8) {
    const float* hp = h + (long)r * 128;
    float x0 = hp[lane], x1 = hp[32 + lane], x2 = hp[64 + lane], x3 = hp[96 + lane];
    float s = wave_sum(x0 * w0 + x1 * w1 + x2 * w2 + x3 * w3);
    float sig = 1.f / (1.f + __expf(-(s + b)));
    a0 += x0 * sig; a1 += x1 * sig; a2 += x2 * sig; a3 += x3 * sig;
  }
  atomicAdd(&g[lane], a0); atomicAdd(&g[32 + lane], a1);
  atomicAdd(&g[64 + lane], a2); atomicAdd(&g[96 + lane], a3);
}

__global__ void missing_kernel(const float* __restrict__ g, const float* __restrict__ w1,
                               const float* __restrict__ b1, const float* __restrict__ w2,
                               const float* __restrict__ b2, float* __restrict__ out) {
  __shared__ float sg[128];
  __shared__ float sh[128];
  int t = threadIdx.x;
  sg[t] = g[t];
  __syncthreads();
  float s = 0.f;
  for (int k = 0; k < 128; k++) s += w1[t * 128 + k] * sg[k];
  sh[t] = fmaxf(s + b1[t], 0.f) * w2[t];
  __syncthreads();
  if (t == 0) {
    float tot = 0.f;
    for (int k = 0; k < 128; k++) tot += sh[k];
    tot += b2[0];
    out[0] = (tot > 20.f) ? tot : log1pf(__expf(tot));
  }
}

// ---------------------------------------------------------------- pair MLP
__global__ void pair_gather_kernel(const float* __restrict__ h0, const float* __restrict__ h1,
                                   const int* __restrict__ cand, _Float16* __restrict__ pairf,
                                   int c0, int cn) {
  long idx = (long)blockIdx.x * 256 + threadIdx.x;
  if (idx >= (long)cn * 256) return;
  int cl = (int)(idx >> 8), col = (int)(idx & 255);
  int c = c0 + cl;
  float v = (col < 128) ? h0[(long)cand[c * 2] * 128 + col]
                        : h1[(long)cand[c * 2 + 1] * 128 + (col - 128)];
  pairf[(long)cl * 256 + col] = (_Float16)v;
}

__global__ __launch_bounds__(256) void pair_final_kernel(
    const float* __restrict__ hidden, const float* __restrict__ lp_w2,
    const float* __restrict__ lp_b2, const float* __restrict__ ewp_w2,
    const float* __restrict__ ewp_b2, float* __restrict__ link, float* __restrict__ wt,
    int c0, int cn) {
  int cl = blockIdx.x * 8 + (threadIdx.x >> 5);
  int lane = threadIdx.x & 31;
  if (cl >= cn) return;
  const float* hp = hidden + (long)cl * 256;
  float s1 = hp[lane] * lp_w2[lane] + hp[32 + lane] * lp_w2[32 + lane]
           + hp[64 + lane] * lp_w2[64 + lane] + hp[96 + lane] * lp_w2[96 + lane];
  s1 = wave_sum(s1);
  float s2 = hp[128 + lane] * ewp_w2[lane] + hp[160 + lane] * ewp_w2[32 + lane]
           + hp[192 + lane] * ewp_w2[64 + lane] + hp[224 + lane] * ewp_w2[96 + lane];
  s2 = wave_sum(s2);
  if (lane == 0) {
    link[c0 + cl] = 1.f / (1.f + __expf(-(s1 + lp_b2[0])));
    wt[c0 + cl]   = 1.f / (1.f + __expf(-(s2 + ewp_b2[0])));
  }
}

// ---------------------------------------------------------------- host
static inline size_t alignup(size_t x) { return (x + 255) & ~(size_t)255; }

extern "C" void kernel_launch(void* const* d_in, const int* in_sizes, int n_in,
                              void* d_out, int out_size, void* d_ws, size_t ws_size,
                              hipStream_t stream) {
  (void)in_sizes; (void)n_in; (void)out_size; (void)ws_size;
  const float* x_room   = (const float*)d_in[0];
  const float* x_portal = (const float*)d_in[1];
  const int*   ei_rp    = (const int*)d_in[2];
  const int*   ei_pr    = (const int*)d_in[3];
  const int*   cand     = (const int*)d_in[4];
  const float* enc_w = (const float*)d_in[5];
  const float* enc_b = (const float*)d_in[6];
  const float* k_w = (const float*)d_in[7];
  const float* k_b = (const float*)d_in[8];
  const float* q_w = (const float*)d_in[9];
  const float* q_b = (const float*)d_in[10];
  const float* v_w = (const float*)d_in[11];
  const float* v_b = (const float*)d_in[12];
  const float* a_w = (const float*)d_in[13];
  const float* a_b = (const float*)d_in[14];
  const float* skip = (const float*)d_in[15];
  const float* a_rel = (const float*)d_in[16];
  const float* m_rel = (const float*)d_in[17];
  const float* p_rel = (const float*)d_in[18];
  const float* cls_w = (const float*)d_in[19];
  const float* cls_b = (const float*)d_in[20];
  const float* attn_w = (const float*)d_in[21];
  const float* attn_b = (const float*)d_in[22];
  const float* ne_w1 = (const float*)d_in[23];
  const float* ne_b1 = (const float*)d_in[24];
  const float* ne_w2 = (const float*)d_in[25];
  const float* ne_b2 = (const float*)d_in[26];
  const float* lp_w1 = (const float*)d_in[27];
  const float* lp_b1 = (const float*)d_in[28];
  const float* lp_w2 = (const float*)d_in[29];
  const float* lp_b2 = (const float*)d_in[30];
  const float* ewp_w1 = (const float*)d_in[31];
  const float* ewp_b1 = (const float*)d_in[32];
  const float* ewp_w2 = (const float*)d_in[33];
  const float* ewp_b2 = (const float*)d_in[34];

  float* out = (float*)d_out;
  float* out_preds   = out;               // 100000*16
  float* out_link    = out + 1600000;     // 200000
  float* out_wt      = out + 1800000;     // 200000
  float* out_missing = out + 2000000;     // 1
  float* out_h0      = out + 2000001;     // 100000*128
  float* out_h1      = out + 14800001;    // 100000*128
  float* out_g       = out + 27600001;    // 128

  // workspace carving
  char* wp = (char*)d_ws;
  auto carve = [&](size_t bytes) { char* r = wp; wp += alignup(bytes); return r; };
  const size_t HBYTES = (size_t)NROOM * 128 * sizeof(float);
  _Float16* hf16[2] = { (_Float16*)carve(HBYTES / 2), (_Float16*)carve(HBYTES / 2) };
  float* hA[2] = { (float*)carve(HBYTES), (float*)carve(HBYTES) };
  float* hB[2] = { (float*)carve(HBYTES), (float*)carve(HBYTES) };
  float* qbuf[2] = { (float*)carve(HBYTES), (float*)carve(HBYTES) };
  float* krel[2] = { (float*)carve(HBYTES), (float*)carve(HBYTES) };
  float* vrel[2] = { (float*)carve(HBYTES), (float*)carve(HBYTES) };
  float* logits = (float*)carve((size_t)NEDGE * 4 * sizeof(float));
  unsigned* nmax = (unsigned*)carve((size_t)NROOM * 4 * sizeof(unsigned));
  float* denom = (float*)carve((size_t)NROOM * 4 * sizeof(float));
  float* aggr = (float*)carve(HBYTES);
  _Float16* pairf = (_Float16*)carve((size_t)CHUNK * 256 * sizeof(_Float16));
  float* hidden = (float*)carve((size_t)CHUNK * 256 * sizeof(float));
  _Float16 *Btq[2][2], *Btk[2][2], *Btv[2][2], *Bta[2][2];
  float *fkb[2][2], *fvb[2][2];
  for (int l = 0; l < 2; l++)
    for (int t = 0; t < 2; t++) {
      Btq[l][t] = (_Float16*)carve(128 * 128 * 2);
      Btk[l][t] = (_Float16*)carve(128 * 128 * 2);
      Btv[l][t] = (_Float16*)carve(128 * 128 * 2);
      Bta[l][t] = (_Float16*)carve(128 * 128 * 2);
      fkb[l][t] = (float*)carve(128 * 4);
      fvb[l][t] = (float*)carve(128 * 4);
    }
  _Float16* Bpair = (_Float16*)carve(256 * 256 * 2);
  float* pb1 = (float*)carve(256 * 4);
  _Float16* Bcls = (_Float16*)carve(128 * 16 * 2);

  // ---- weight prep (data-independent)
  for (int l = 0; l < 2; l++)
    for (int t = 0; t < 2; t++) {
      long wo = (long)(l * 2 + t) * 128 * 128;
      long bo = (long)(l * 2 + t) * 128;
      long ro = (long)(l * 2 + t) * 4 * 32 * 32;
      prep_weight_plain<<<64, 256, 0, stream>>>(q_w + wo, Btq[l][t], 128, 128);
      prep_weight_fused<<<64, 256, 0, stream>>>(k_w + wo, k_b + bo, a_rel + ro, Btk[l][t], fkb[l][t]);
      prep_weight_fused<<<64, 256, 0, stream>>>(v_w + wo, v_b + bo, m_rel + ro, Btv[l][t], fvb[l][t]);
      prep_weight_plain<<<64, 256, 0, stream>>>(a_w + wo, Bta[l][t], 128, 128);
    }
  prep_weight_plain<<<8, 256, 0, stream>>>(cls_w, Bcls, 128, 16);
  prep_weight_pair<<<256, 256, 0, stream>>>(lp_w1, ewp_w1, lp_b1, ewp_b1, Bpair, pb1);

  // ---- encoders
  encode_kernel<<<50000, 256, 0, stream>>>(x_room, enc_w, enc_b, hA[0], NROOM);
  encode_kernel<<<50000, 256, 0, stream>>>(x_portal, enc_w + 128, enc_b + 128, hA[1], NPORTAL);

  const dim3 gridM((NROOM + 127) / 128, 1);

  // ---- HGT layers
  float* hcur[2] = { hA[0], hA[1] };
  for (int l = 0; l < 2; l++) {
    float* hout[2];
    if (l == 0) { hout[0] = hB[0]; hout[1] = hB[1]; }
    else        { hout[0] = out_h0; hout[1] = out_h1; }
    for (int t = 0; t < 2; t++) {
      convert_kernel<<<50000, 256, 0, stream>>>(hcur[t], hf16[t], (long)NROOM * 128, 0);
      gemm_f16_kernel<8, 4><<<gridM, 256, 0, stream>>>(hf16[t], Btq[l][t],
          q_b + (long)(l * 2 + t) * 128, qbuf[t], NROOM, 128, 0, nullptr, nullptr);
      gemm_f16_kernel<8, 4><<<gridM, 256, 0, stream>>>(hf16[t], Btk[l][t],
          fkb[l][t], krel[t], NROOM, 128, 0, nullptr, nullptr);
      gemm_f16_kernel<8, 4><<<gridM, 256, 0, stream>>>(hf16[t], Btv[l][t],
          fvb[l][t], vrel[t], NROOM, 128, 0, nullptr, nullptr);
    }
    for (int dst_t = 0; dst_t < 2; dst_t++) {
      int et = 1 - dst_t;           // dst_t=1 -> et=0 (room->portal); dst_t=0 -> et=1
      int src_t = et;               // src type == et
      const int* ei = (et == 0) ? ei_rp : ei_pr;
      fill_u32_kernel<<<1563, 256, 0, stream>>>(nmax, 0x007FFFFFu, (long)NROOM * 4);   // enc(-inf)
      fill_u32_kernel<<<1563, 256, 0, stream>>>((unsigned*)denom, 0u, (long)NROOM * 4);
      fill_u32_kernel<<<50000, 256, 0, stream>>>((unsigned*)aggr, 0u, (long)NROOM * 128);
      edge_logits_kernel<<<NEDGE / 8, 256, 0, stream>>>(qbuf[dst_t], krel[src_t], ei,
                                                        p_rel + (l * 2 + et) * 4, logits, nmax, NEDGE);
      edge_exp_kernel<<<(NEDGE * 4) / 256, 256, 0, stream>>>(logits, nmax, denom, ei, NEDGE);
      edge_aggr_kernel<<<NEDGE / 8, 256, 0, stream>>>(logits, denom, vrel[src_t], ei, aggr, NEDGE);
      convert_kernel<<<50000, 256, 0, stream>>>(aggr, hf16[dst_t], (long)NROOM * 128, 1);  // gelu
      gemm_f16_kernel<8, 4><<<gridM, 256, 0, stream>>>(hf16[dst_t], Bta[l][dst_t],
          a_b + (long)(l * 2 + dst_t) * 128, hout[dst_t], NROOM, 128, 1,
          hcur[dst_t], skip + (l * 2 + dst_t));
    }
    hcur[0] = hout[0]; hcur[1] = hout[1];
  }

  // ---- node classifier (N=16 -> single tile)
  convert_kernel<<<50000, 256, 0, stream>>>(out_h0, hf16[0], (long)NROOM * 128, 0);
  gemm_f16_kernel<1, 4><<<gridM, 256, 0, stream>>>(hf16[0], Bcls, cls_b, out_preds,
                                                   NROOM, 16, 0, nullptr, nullptr);

  // ---- attention pooling + missing head
  fill_u32_kernel<<<1, 128, 0, stream>>>((unsigned*)out_g, 0u, 128);
  pool_kernel<<<512, 256, 0, stream>>>(out_h0, attn_w, attn_b, out_g, NROOM);
  pool_kernel<<<512, 256, 0, stream>>>(out_h1, attn_w, attn_b, out_g, NPORTAL);
  missing_kernel<<<1, 128, 0, stream>>>(out_g, ne_w1, ne_b1, ne_w2, ne_b2, out_missing);

  // ---- link / weight prediction over candidate pairs (chunked)
  const dim3 gridP((CHUNK + 127) / 128, 2);   // N=256 -> two 128-col chunks
  for (int c0 = 0; c0 < NCAND; c0 += CHUNK) {
    pair_gather_kernel<<<(CHUNK * 256) / 256, 256, 0, stream>>>(out_h0, out_h1, cand, pairf, c0, CHUNK);
    gemm_f16_kernel<8, 8><<<gridP, 256, 0, stream>>>(pairf, Bpair, pb1, hidden,
                                                     CHUNK, 256, 2, nullptr, nullptr);
    pair_final_kernel<<<CHUNK / 8, 256, 0, stream>>>(hidden, lp_w2, lp_b2, ewp_w2, ewp_b2,
                                                     out_link, out_wt, c0, CHUNK);
  }
}
